// LSTMModel_4174708212151
// MI455X (gfx1250) — matrix operand order repaired and checked by reading, not compiled, not run on hardware
//
#include <hip/hip_runtime.h>
#include <hip/hip_bf16.h>

typedef __attribute__((ext_vector_type(16))) __bf16 v16bf;
typedef __attribute__((ext_vector_type(8)))  __bf16 v8bf;
typedef __attribute__((ext_vector_type(8)))  float  v8f;
typedef __attribute__((ext_vector_type(4)))  unsigned u32x4;
typedef __attribute__((ext_vector_type(8)))  unsigned u32x8;

// Problem sizes
constexpr int Bb = 64, Tt = 2048, Dd = 128, Uu = 256, G4 = 1024;
// Partitioning: 2 clusters x 16 WGs; each WG owns 16 units (64 gate columns)
constexpr int NCLU = 2, NWG = 16;
constexpr int MB  = Bb / NCLU;   // 32 batch rows per cluster
constexpr int UPW = Uu / NWG;    // 16 units per WG
constexpr int KH  = Uu;          // 256 (recurrent K)
constexpr int KX  = Dd;          // 128 (input K)
constexpr int KT  = KH + KX;     // 384 combined K
constexpr int NKB = KT / 32;     // 12 k-blocks of 32
// LDS strides padded (+8 bf16) so fragment lanes hit distinct banks
constexpr int WSTR = KT + 8;     // 392
constexpr int HSTR = KH + 8;     // 264
constexpr int XSTR = KX + 8;     // 136

__device__ inline __bf16 f2bf(float f) {
  unsigned u = __builtin_bit_cast(unsigned, f);
  unsigned r = (u + 0x7FFFu + ((u >> 16) & 1u)) >> 16;   // round-to-nearest-even
  return __builtin_bit_cast(__bf16, (unsigned short)r);
}

// Issue a TDM 2D tile load: inputs[clu*MB .. +31][t][0..127] (f32) -> LDS staging.
// D# per CDNA5 ISA. Wave-level issue (TDM ignores EXEC).
__device__ inline void tdm_load_x(const float* inp, unsigned ldsOff, int clu, int t) {
  unsigned long long ga =
      (unsigned long long)(inp + ((size_t)(clu * MB) * Tt + t) * Dd);
  u32x4 g0;
  g0[0] = 1u;                                              // count=1, user mode
  g0[1] = ldsOff;                                          // lds_addr (bytes)
  g0[2] = (unsigned)ga;                                    // global_addr[31:0]
  g0[3] = (unsigned)((ga >> 32) & 0x1FFFFFFull)            // global_addr[56:32]
          | 0x80000000u;                                   // type=2 ("image")
  u32x8 g1;
  g1[0] = 2u << 16;                 // data_size=2 -> 4-byte elements
  g1[1] = (unsigned)Dd << 16;       // tensor_dim0 = 128   (bits [79:48] low half)
  g1[2] = (unsigned)MB << 16;       // tensor_dim1 = 32    (bits [111:80] low half)
  g1[3] = (unsigned)Dd << 16;       // tile_dim0 = 128     (bits [127:112])
  g1[4] = (unsigned)MB;             // tile_dim1 = 32      (bits [143:128])
  g1[5] = (unsigned)(Tt * Dd);      // tensor_dim0_stride = 262144 elems (low 32)
  g1[6] = 0u;                       // stride high bits / dim1_stride low
  g1[7] = 0u;
  u32x4 gz = {0u, 0u, 0u, 0u};      // groups 2/3: unused dims (tile_dim3=0)
  asm volatile("tensor_load_to_lds %0, %1, %2, %3"
               :: "s"(g0), "s"(g1), "s"(gz), "s"(gz)
               : "memory");
}

__global__ __launch_bounds__(256, 1)
void lstm_cluster_kernel(const float* __restrict__ inp,   // [B,T,D]
                         const float* __restrict__ Wx,    // [D,4U]
                         const float* __restrict__ Wh,    // [U,4U]
                         const float* __restrict__ bias,  // [4U]
                         float* __restrict__ out,         // [B,T,U]
                         __bf16* __restrict__ hbuf)       // ws: [2][B][U] bf16 double buffer
{
  __shared__ __bf16 sW[64 * WSTR];   // [col 0..63][k 0..383] = [Wh;Wx] slice, K-major
  __shared__ __bf16 sH[MB * HSTR];   // h  [32][256] bf16 row-major
  __shared__ __bf16 sX[MB * XSTR];   // x_t[32][128] bf16 row-major (padded)
  __shared__ float  sXf[MB * Dd];    // TDM staging: x tile f32 [32][128] packed
  __shared__ float  sZ[4 * MB * 17]; // staged gate pre-activations [gate][row][unit]
  __shared__ float  sC[MB * UPW];    // cell state for owned units
  __shared__ float  sB[64];          // bias for owned 64 gate columns

  const int tid  = threadIdx.x;
  const int lane = tid & 31;
  const int wv   = tid >> 5;
  const int mt   = wv & 1;           // 2 M-tiles (32 rows)
  const int nt   = wv >> 1;          // 4 N-tiles == 4 gates (16 units each)
  const int wg   = blockIdx.x & (NWG - 1);
  const int clu  = blockIdx.x / NWG;
  const unsigned sXfOff = (unsigned)(size_t)(&sXf[0]);

  // ---------------- init: weights/bias/state into LDS ----------------
  if (wv == 0) tdm_load_x(inp, sXfOff, clu, 0);           // x_0 via TDM
  for (int i = tid; i < 64 * KT; i += 256) {
    int col = i / KT, k = i - col * KT;
    int gate = col >> 4, u = col & 15;
    int gcol = gate * Uu + wg * UPW + u;                  // column in [0,1024)
    float v = (k < KH) ? Wh[k * G4 + gcol] : Wx[(k - KH) * G4 + gcol];
    sW[col * WSTR + k] = f2bf(v);
  }
  if (tid < 64) {
    int gate = tid >> 4, u = tid & 15;
    sB[tid] = bias[gate * Uu + wg * UPW + u];
  }
  for (int i = tid; i < MB * UPW; i += 256) sC[i] = 0.0f;
  for (int i = tid; i < MB * HSTR; i += 256) sH[i] = f2bf(0.0f);
  if (wv == 0) __builtin_amdgcn_s_wait_tensorcnt(0);
  __syncthreads();
  for (int i = tid; i < MB * Dd; i += 256)                // convert x_0 -> bf16
    sX[(i >> 7) * XSTR + (i & 127)] = f2bf(sXf[i]);
  __syncthreads();

  // Per-lane fragment geometry (ISA 7.12.2 VGPR layouts, wave32)
  const int m_row = mt * 16 + (lane & 15);            // A: row
  const int koff  = (lane >> 4) << 3;                 // A: +0 / +8 within k-block
  const int khalf = (lane >> 4) << 4;                 // unused; B uses koff (+0 / +8) like A
  const int ncol  = lane & 15;                        // B/D: column
  const __bf16* wcol = sW + (nt * 16 + ncol) * WSTR;

  for (int t = 0; t < Tt; ++t) {
    // ---------------- GEMM: z = [h|x_t] @ [Wh;Wx](:,owned cols) ----------------
    v8f acc = {};
#pragma unroll
    for (int kb = 0; kb < NKB; ++kb) {
      const __bf16* abase = (kb < 8) ? (sH + m_row * HSTR + kb * 32)
                                     : (sX + m_row * XSTR + (kb - 8) * 32);
      v8bf a0 = *(const v8bf*)(abase + koff);         // K runs of 8 contiguous bf16
      v8bf a1 = *(const v8bf*)(abase + 16 + koff);
      v16bf a = __builtin_shufflevector(a0, a1, 0,1,2,3,4,5,6,7,8,9,10,11,12,13,14,15);
      const __bf16* bbase = wcol + kb * 32 + koff;    // K runs of 8 contiguous bf16 (K-major W)
      v8bf b0 = *(const v8bf*)(bbase);
      v8bf b1 = *(const v8bf*)(bbase + 16);
      v16bf b = __builtin_shufflevector(b0, b1, 0,1,2,3,4,5,6,7,8,9,10,11,12,13,14,15);
      acc = __builtin_amdgcn_wmma_f32_16x16x32_bf16(false, a, false, b,
                                                    (short)0, acc, false, false);
    }
    // stage z into LDS (D layout: vgpr r -> M = 8*(lane>=16)+r, N = lane&15)
    {
      int mb = mt * 16 + ((lane >> 4) << 3);
#pragma unroll
      for (int r = 0; r < 8; ++r) sZ[(nt * MB + mb + r) * 17 + ncol] = acc[r];
    }
    __syncthreads();   // sync1: sZ ready; sX/sXf reads of this step done

    // kick TDM for x_{t+1} into staging (idle since last convert)
    if (wv == 0) tdm_load_x(inp, sXfOff, clu, (t + 1 < Tt) ? t + 1 : t);

    // ---------------- gates, state update, output ----------------
    __bf16* wb = hbuf + (size_t)((t + 1) & 1) * Bb * Uu;
    for (int cell = tid; cell < MB * UPW; cell += 256) {   // 2 iters
      int b = cell >> 4, u = cell & 15;
      float zi = sZ[(0 * MB + b) * 17 + u] + sB[u];
      float zf = sZ[(1 * MB + b) * 17 + u] + sB[16 + u];
      float zg = sZ[(2 * MB + b) * 17 + u] + sB[32 + u];
      float zo = sZ[(3 * MB + b) * 17 + u] + sB[48 + u];
      float ig = 1.0f / (1.0f + __expf(-zi));
      float fg = 1.0f / (1.0f + __expf(-zf));
      float og = 1.0f / (1.0f + __expf(-zo));
      float gg = 2.0f / (1.0f + __expf(-2.0f * zg)) - 1.0f;   // tanh
      float c  = fg * sC[cell] + ig * gg;
      sC[cell] = c;
      float h  = og * (2.0f / (1.0f + __expf(-2.0f * c)) - 1.0f);
      int gb = clu * MB + b;
      wb[(size_t)gb * Uu + wg * UPW + u] = f2bf(h);          // publish h slice
      out[((size_t)gb * Tt + t) * Uu + wg * UPW + u] = h;
    }
    if (tid < MB)  // warm L2 for x_{t+2}
      __builtin_prefetch(inp + ((size_t)(clu * MB + tid) * Tt +
                                (t + 2 < Tt ? t + 2 : t)) * Dd, 0, 1);

    if (wv == 0) __builtin_amdgcn_s_wait_tensorcnt(0);  // x_{t+1} staged
    __threadfence();                       // make hbuf stores device-visible
    __syncthreads();                       // sync2: hbuf published, TDM complete
    __builtin_amdgcn_s_cluster_barrier();  // SE-local cluster h-exchange point

    // async-copy full h_{t+1} (all 16 WGs' slices) global -> LDS (ASYNCcnt)
    {
      unsigned long long hb =
          (unsigned long long)(hbuf + (size_t)((t + 1) & 1) * Bb * Uu);
      for (int i = tid; i < MB * 32; i += 256) {       // 4 x 16B chunks / thread
        int row = i >> 5, j = i & 31;
        unsigned ldsOff = (unsigned)(size_t)(sH + row * HSTR + j * 8);
        unsigned voff   = (unsigned)(((clu * MB + row) * 32 + j) * 16);
        asm volatile("global_load_async_to_lds_b128 %0, %1, %2"
                     :: "v"(ldsOff), "v"(voff), "s"(hb) : "memory");
      }
    }
    // overlap: convert staged x_{t+1} f32 -> bf16 while async loads fly
    for (int i = tid; i < MB * Dd; i += 256)           // 16 elems / thread
      sX[(i >> 7) * XSTR + (i & 127)] = f2bf(sXf[i]);

    asm volatile("s_wait_asynccnt 0" ::: "memory");    // this wave's h chunks landed
    __syncthreads();                                   // sync3: sH/sX ready for t+1
  }
}

extern "C" void kernel_launch(void* const* d_in, const int* in_sizes, int n_in,
                              void* d_out, int out_size, void* d_ws, size_t ws_size,
                              hipStream_t stream) {
  const float* inp  = (const float*)d_in[0];   // [64,2048,128]
  const float* Wx   = (const float*)d_in[1];   // [128,1024]
  const float* Wh   = (const float*)d_in[2];   // [256,1024]
  const float* bias = (const float*)d_in[3];   // [1024]
  float* out = (float*)d_out;                  // [64,2048,256]
  __bf16* hbuf = (__bf16*)d_ws;                // needs 2*64*256*2 = 64 KB

  (void)in_sizes; (void)n_in; (void)out_size; (void)ws_size;
  lstm_cluster_kernel<<<dim3(NCLU * NWG), dim3(256), 0, stream>>>(
      inp, Wx, Wh, bias, out, hbuf);
}
